// FFTNetwork_40338332844581
// MI455X (gfx1250) — compile-verified, matrix operand order checked by reading, not run on hardware
//
#include <hip/hip_runtime.h>
#include <hip/hip_bf16.h>

typedef __attribute__((ext_vector_type(16))) _Float16 v16h;
typedef __attribute__((ext_vector_type(8)))  _Float16 v8h;
typedef __attribute__((ext_vector_type(8)))  float    v8f;

#define MODN 26
#define NN   64
#define BB   1024
#define ROWS (NN*NN)            // 4096 key rows
#define TWOPI_26 0.2417482132f  // 2*pi/26

// ---------------------------------------------------------------------------
// K1: softmax over each of 4096 rows of key_param (len 26).
// Writes f32 key_probs to the output tail and an f16, K-padded-to-32 A-matrix
// for the WMMA DFT GEMM.
// ---------------------------------------------------------------------------
__global__ void k1_softmax(const float* __restrict__ key_param,
                           float* __restrict__ keyp_out,
                           _Float16* __restrict__ Akp) {
    int row = blockIdx.x * blockDim.x + threadIdx.x;
    if (row >= ROWS) return;
    const float* x = key_param + row * MODN;
    float m = -1e30f;
    #pragma unroll
    for (int k = 0; k < MODN; ++k) m = fmaxf(m, x[k]);
    float e[MODN]; float s = 0.f;
    #pragma unroll
    for (int k = 0; k < MODN; ++k) { e[k] = __expf(x[k] - m); s += e[k]; }
    float inv = 1.f / s;
    float*    ko = keyp_out + row * MODN;
    _Float16* ao = Akp + row * 32;
    #pragma unroll
    for (int k = 0; k < MODN; ++k) {
        float p = e[k] * inv;
        ko[k] = p;
        ao[k] = (_Float16)p;
    }
    #pragma unroll
    for (int k = MODN; k < 32; ++k) ao[k] = (_Float16)0.f;
}

// ---------------------------------------------------------------------------
// K2: G = Akp (4096x32 f16) x DFT (32x64 f16), f32 accumulate via WMMA.
// Output G[row][j]: j<26 -> Re(DFT), 26<=j<52 -> Im(DFT) (= -sin sum), rest pad.
// One 16x16 output tile per wave, single v_wmma_f32_16x16x32_f16 each.
// ---------------------------------------------------------------------------
__global__ void k2_dft_gemm(const _Float16* __restrict__ Akp,
                            float* __restrict__ G) {
    int tid   = threadIdx.x;
    int wave  = tid >> 5;          // 0..3  -> N-tile
    int lane  = tid & 31;
    int mtile = blockIdx.x;        // 0..255
    int half  = lane >> 4;
    int mn    = lane & 15;

    // A fragment: lane covers row (mtile*16+mn); two contiguous 128b chunks:
    // k in [half*8, half*8+8) and [16+half*8, 16+half*8+8)
    const v8h* arow = (const v8h*)(Akp + (size_t)(mtile * 16 + mn) * 32);
    v8h lo = arow[half];
    v8h hi = arow[2 + half];
    v16h a;
    #pragma unroll
    for (int e = 0; e < 8; ++e) { a[e] = lo[e]; a[8 + e] = hi[e]; }

    // B fragment: column j = wave*16 + mn, k from the pair-striping formula.
    v16h b;
    #pragma unroll
    for (int e = 0; e < 16; ++e) {
        int k = ((e & 8) << 1) + (half << 3) + (e & 7);   // 0..31
        int j = wave * 16 + mn;                           // 0..63
        float v = 0.f;
        if (k < MODN) {
            if (j < MODN) {                               // Re: cos(2*pi*j*k/26)
                int mm = (j * k) % MODN;
                v = __cosf((float)mm * TWOPI_26);
            } else if (j < 2 * MODN) {                    // Im: -sin(2*pi*g*k/26)
                int g  = j - MODN;
                int mm = (g * k) % MODN;
                v = -__sinf((float)mm * TWOPI_26);
            }
        }
        b[e] = (_Float16)v;
    }

    v8f c = {};
    c = __builtin_amdgcn_wmma_f32_16x16x32_f16(false, a, false, b,
                                               (short)0, c, false, false);

    // D: VGPR r -> row r+half*8, col = mn (per 16x16 f32 C/D layout)
    float* gbase = G + (size_t)(mtile * 16) * 64 + wave * 16;
    #pragma unroll
    for (int r = 0; r < 8; ++r)
        gbase[(size_t)(r + half * 8) * 64 + mn] = c[r];
}

// ---------------------------------------------------------------------------
// K3: per-(b,i,f) complex product over l of G[l,i, (f*P[b,l]) mod 26].
// LDS-staged version: block = (i, 64-wide b-group). The full G slice for this
// i (64 l-rows x 64 floats = 16 KB) and the P sub-block (64 b x 64 l ints =
// 16 KB) are loaded once with coalesced b128 loads; the 64-step chain then
// runs entirely out of LDS. Global G traffic: ~470 MB -> ~16 MB.
// Threads: 896 = 64 b_local x 14 f. Writes X (f16, 32-wide rows):
// cols 0..13 = Re(f), 14..27 = Im(f), 28..31 = 0.
// ---------------------------------------------------------------------------
__global__ void k3_chain(const int* __restrict__ P,
                         const float* __restrict__ G,
                         _Float16* __restrict__ Xw) {
    __shared__ float Gl[NN * 64];     // [l][j]   16 KB
    __shared__ int   Pl[NN * NN];     // [bl][l]  16 KB

    int tid    = threadIdx.x;          // 0..895
    int i      = blockIdx.x >> 4;      // 0..63
    int bgroup = blockIdx.x & 15;      // 0..15

    // Stage G slice: 4096 floats = 1024 float4 chunks.
    for (int idx = tid; idx < 1024; idx += 896) {
        int l = idx >> 4, ch = idx & 15;
        ((float4*)Gl)[idx] =
            *(const float4*)(G + ((size_t)(l * NN + i) * 64 + ch * 4));
    }
    // Stage P sub-block: rows b = bgroup*64 + bl, 1024 int4 chunks.
    for (int idx = tid; idx < 1024; idx += 896) {
        int bl = idx >> 4, ch = idx & 15;
        ((int4*)Pl)[idx] =
            *(const int4*)(P + ((size_t)(bgroup * NN + bl) * NN + ch * 4));
    }
    __syncthreads();

    int bl = tid / 14;     // 0..63
    int f  = tid % 14;     // 0..13
    float ar = 1.f, ai = 0.f;
    #pragma unroll 4
    for (int l = 0; l < NN; ++l) {
        int p = Pl[bl * NN + l];
        int g = (f * p) % MODN;
        float br = Gl[l * 64 + g];
        float bi = Gl[l * 64 + MODN + g];
        float nr = ar * br - ai * bi;
        float ni = ar * bi + ai * br;
        ar = nr; ai = ni;
    }
    int b = bgroup * NN + bl;
    _Float16* xr = Xw + (size_t)(b * NN + i) * 32;
    xr[f]      = (_Float16)ar;
    xr[14 + f] = (_Float16)ai;
    if (f < 4) xr[28 + f] = (_Float16)0.f;
}

// ---------------------------------------------------------------------------
// K4: inverse rDFT GEMM (65536x28 f16) x W(28x26 f16) via WMMA (2 N-tiles),
// then per-row clip / normalize / log into d_out. One 16-row strip per wave;
// epilogue uses all 32 lanes (13 cols per half-lane, sum via shfl_xor 16).
// ---------------------------------------------------------------------------
__global__ void k4_idft(const _Float16* __restrict__ Xw,
                        float* __restrict__ out_logp) {
    __shared__ float stage[4][16][32];
    int tid   = threadIdx.x;
    int wave  = tid >> 5;
    int lane  = tid & 31;
    int strip = blockIdx.x * 4 + wave;   // 0..4095 (16 chain-rows each)
    int half  = lane >> 4;
    int mn    = lane & 15;

    const v8h* arow = (const v8h*)(Xw + (size_t)(strip * 16 + mn) * 32);
    v8h lo = arow[half];
    v8h hi = arow[2 + half];
    v16h a;
    #pragma unroll
    for (int e = 0; e < 8; ++e) { a[e] = lo[e]; a[8 + e] = hi[e]; }

    #pragma unroll
    for (int nt = 0; nt < 2; ++nt) {
        v16h b;
        #pragma unroll
        for (int e = 0; e < 16; ++e) {
            int k = ((e & 8) << 1) + (half << 3) + (e & 7); // 0..31
            int c = nt * 16 + mn;                            // 0..31
            float v = 0.f;
            if (c < MODN && k < 28) {
                if (k < 14) {                 // Re(f) coefficients
                    int f = k;
                    if (f == 0)       v = 1.f / 26.f;
                    else if (f == 13) v = ((c & 1) ? -1.f : 1.f) / 26.f;
                    else {
                        int mm = (f * c) % MODN;
                        v = (2.f / 26.f) * __cosf((float)mm * TWOPI_26);
                    }
                } else {                      // Im(f) coefficients
                    int f = k - 14;
                    if (f >= 1 && f <= 12) {
                        int mm = (f * c) % MODN;
                        v = -(2.f / 26.f) * __sinf((float)mm * TWOPI_26);
                    }
                }
            }
            b[e] = (_Float16)v;
        }
        v8f d = {};
        d = __builtin_amdgcn_wmma_f32_16x16x32_f16(false, a, false, b,
                                                   (short)0, d, false, false);
        #pragma unroll
        for (int r = 0; r < 8; ++r)
            stage[wave][r + half * 8][nt * 16 + mn] = d[r];
    }
    __syncthreads();

    // Epilogue: row = lane&15, column half = lane>>4 (13 cols each).
    {
        int r = lane & 15;
        int h = lane >> 4;
        float part[13]; float s = 0.f;
        #pragma unroll
        for (int cc = 0; cc < 13; ++cc) {
            float x = stage[wave][r][h * 13 + cc];
            x = (x < 0.f) ? 0.f : x;
            part[cc] = x; s += x;
        }
        s += __shfl_xor(s, 16);         // combine halves (wave32 permute)
        if (s == 0.f) s = 1.f;          // reference: where(s==0, 1, s)
        float inv = 1.f / s;
        size_t rowg = (size_t)strip * 16 + r;   // = b*64 + i
        float* o = out_logp + rowg * MODN + h * 13;
        #pragma unroll
        for (int cc = 0; cc < 13; ++cc)
            o[cc] = __logf(part[cc] * inv + 1e-12f);
    }
}

// ---------------------------------------------------------------------------
extern "C" void kernel_launch(void* const* d_in, const int* in_sizes, int n_in,
                              void* d_out, int out_size, void* d_ws, size_t ws_size,
                              hipStream_t stream) {
    const int*   P         = (const int*)d_in[0];    // (B, N) int
    const float* key_param = (const float*)d_in[1];  // (N*N, 26) f32

    float* out      = (float*)d_out;
    float* out_logp = out;                                   // B*N*26
    float* out_keyp = out + (size_t)BB * NN * MODN;          // N*N*26

    char* ws = (char*)d_ws;
    _Float16* Akp = (_Float16*)ws;                           // 4096*32*2  = 256 KB
    float*    G   = (float*)(ws + 256 * 1024);               // 4096*64*4  =   1 MB
    _Float16* Xw  = (_Float16*)(ws + 256 * 1024 + 1024 * 1024); // 65536*32*2 = 4 MB

    k1_softmax <<<ROWS / 256, 256, 0, stream>>>(key_param, out_keyp, Akp);
    k2_dft_gemm<<<ROWS / 16, 128, 0, stream>>>(Akp, G);
    k3_chain   <<<BB, 896, 0, stream>>>(P, G, Xw);          // 64 i x 16 bgroups
    k4_idft    <<<(BB * NN) / 64, 128, 0, stream>>>(Xw, out_logp);
}
